// TimeAwareAttention_80539226735120
// MI455X (gfx1250) — compile-verified, matrix operand order checked by reading, not run on hardware
//
#include <hip/hip_runtime.h>
#include <math.h>

// Problem constants (match reference)
#define NH 4096    // n_history
#define DE 512     // embed dim
#define HH 8       // time-weight MLP hidden
#define QB 16      // query rows per workgroup
#define KB 128     // keys per iteration (8 waves * 16 keys)
#define DSLICE 64  // d-columns owned per wave for the P@V accumulator

typedef __attribute__((ext_vector_type(16))) _Float16 v16h;
typedef __attribute__((ext_vector_type(8)))  _Float16 v8h;
typedef __attribute__((ext_vector_type(8)))  float    v8f;

static __device__ __forceinline__ v16h ldfrag(const _Float16* p) {
  v16h r;
  ((v8h*)&r)[0] = *(const v8h*)(p);
  ((v8h*)&r)[1] = *(const v8h*)(p + 16);
  return r;
}

// ---------------------------------------------------------------------------
// Pre-pass: f32 -> f16 conversion.  q_h = Q * (1/sqrt(D)) row-major,
// k_h = K row-major, kT_h = K^T (f16, [D][N]) for the P@V B-fragments.
// ---------------------------------------------------------------------------
__global__ __launch_bounds__(256) void taa_convert_kernel(
    const float* __restrict__ Q, const float* __restrict__ K,
    _Float16* __restrict__ qh, _Float16* __restrict__ kh,
    _Float16* __restrict__ kT) {
  const int i = blockIdx.x * blockDim.x + threadIdx.x;   // [0, N*D)
  const float scale = 0.044194173824159216f;             // 1/sqrt(512)
  qh[i] = (_Float16)(Q[i] * scale);
  kh[i] = (_Float16)(K[i]);
  // transposed copy: interpret i as d*N + key -> coalesced write, L2-cached read
  const int key = i & (NH - 1);
  const int d   = i >> 12;                               // i / NH
  kT[i] = (_Float16)(K[(size_t)key * DE + d]);
}

// ---------------------------------------------------------------------------
// Fused time-aware flash attention.
// Workgroup: 16 query rows, 8 wave32s.
//   QK^T : wave w owns keys [kb + 16w, kb + 16w + 16), full-D reduction,
//          A-fragments from LDS-staged Q (async-copied once).
//   P@V  : wave w owns output d-slice [64w, 64w + 64).
//   softmax: wave w owns rows 2w, 2w+1 (shfl-butterfly row reductions).
// ---------------------------------------------------------------------------
__global__ __launch_bounds__(256) void taa_attn_kernel(
    const _Float16* __restrict__ qh, const _Float16* __restrict__ kh,
    const _Float16* __restrict__ kT, const float* __restrict__ times,
    const float* __restrict__ w1, const float* __restrict__ b1,
    const float* __restrict__ w2, const float* __restrict__ b2p,
    float* __restrict__ out) {
  __shared__ __align__(16) _Float16 sQ[QB * DE];   // 16x512 f16 = 16 KB
  __shared__ __align__(16) float    sS[QB * KB];   // 16x128 f32 =  8 KB
  __shared__ __align__(16) _Float16 sP[QB * KB];   // 16x128 f16 =  4 KB
  __shared__ float sAlpha[QB];
  __shared__ float sL[QB];
  __shared__ float sTq[QB];

  const int tid   = threadIdx.x;
  const int lane  = tid & 31;
  const int wave  = tid >> 5;
  const int hrow  = lane & 15;   // M (A) / N (B,C) fragment index
  const int sel   = lane >> 4;   // which K-half of the fragment
  const int qbase = blockIdx.x * QB;
  const int dbase = wave * DSLICE;

  // ---- one-time async stage of the Q block into LDS (64 B per thread) ----
  {
    const _Float16* gq = qh + (size_t)qbase * DE + tid * 32;
    const unsigned  lo = (unsigned)(uintptr_t)(const void*)(&sQ[tid * 32]);
    asm volatile(
        "global_load_async_to_lds_b128 %0, %1, off\n\t"
        "global_load_async_to_lds_b128 %0, %1, off offset:16\n\t"
        "global_load_async_to_lds_b128 %0, %1, off offset:32\n\t"
        "global_load_async_to_lds_b128 %0, %1, off offset:48"
        :: "v"(lo), "v"(gq) : "memory");
    asm volatile("s_wait_asynccnt 0x0" ::: "memory");
  }

  // Tiny MLP parameters -> registers (uniform loads)
  float w1r[HH], b1r[HH], w2r[HH];
#pragma unroll
  for (int j = 0; j < HH; ++j) { w1r[j] = w1[j]; b1r[j] = b1[j]; w2r[j] = w2[j]; }
  const float b2 = b2p[0];

  if (tid < QB) sTq[tid] = times[qbase + tid];

  v8f acc[4] = {};                         // 16 x 64 f32 output slice
  float mrow[2] = {-3.0e38f, -3.0e38f};    // per-owned-row softmax state
  float lrow[2] = {0.0f, 0.0f};
  __syncthreads();                         // sQ (all waves) + sTq visible

  const _Float16* qlds = sQ + hrow * DE + sel * 8;

  for (int kb = 0; kb < NH; kb += KB) {
    // ---- 0) prefetch next key block (K rows + K^T rows) into caches ----
    if (kb + KB < NH) {
      const _Float16* kn =
          kh + (size_t)(kb + KB + wave * 16 + hrow) * DE + sel * 256;
#pragma unroll
      for (int pf = 0; pf < 4; ++pf) __builtin_prefetch(kn + pf * 64, 0, 1);
#pragma unroll
      for (int t = 0; t < 4; ++t) {
        const _Float16* vn =
            kT + (size_t)(dbase + t * 16 + hrow) * NH + kb + KB;
        __builtin_prefetch(vn, 0, 1);
        __builtin_prefetch(vn + 64, 0, 1);
      }
    }

    // ---- 1) S tile: this wave's 16 keys x 16 q-rows over full D ----
    // Register double-buffer: fragments for step kk+1 load during WMMA kk.
    const _Float16* krow = kh + (size_t)(kb + wave * 16 + hrow) * DE + sel * 8;
    v16h aq[2], bk[2];
    aq[0] = ldfrag(qlds);                  // LDS
    bk[0] = ldfrag(krow);                  // global
    v8f sa = {}, sb = {};
#pragma unroll
    for (int kk = 0; kk < 16; ++kk) {
      const int cur = kk & 1;
      if (kk < 15) {
        aq[cur ^ 1] = ldfrag(qlds + (kk + 1) * 32);
        bk[cur ^ 1] = ldfrag(krow + (kk + 1) * 32);
      }
      if (cur)
        sb = __builtin_amdgcn_wmma_f32_16x16x32_f16(false, aq[cur], false,
                                                    bk[cur], (short)0, sb,
                                                    false, false);
      else
        sa = __builtin_amdgcn_wmma_f32_16x16x32_f16(false, aq[cur], false,
                                                    bk[cur], (short)0, sa,
                                                    false, false);
    }
    // store finished S tile (single writer per element; no atomics needed)
#pragma unroll
    for (int v = 0; v < 8; ++v) {
      sS[(v + 8 * sel) * KB + wave * 16 + hrow] = sa[v] + sb[v];
    }
    // Barrier A: sS stores visible; also orders prev-iteration sP reads
    // (PV ds_loads) before this iteration's sP writes below.
    __syncthreads();

    // ---- 2) time-weight MLP + online softmax (wave-parallel rows) ----
#pragma unroll
    for (int rr = 0; rr < 2; ++rr) {
      const int r = 2 * wave + rr;
      const float tq = sTq[r];
      float sv[4];
      float rmax = -3.0e38f;
#pragma unroll
      for (int c4 = 0; c4 < 4; ++c4) {
        const int col = c4 * 32 + lane;
        float s = sS[r * KB + col];
        const float td = fabsf(tq - times[kb + col]);
        float z = b2;
#pragma unroll
        for (int j = 0; j < HH; ++j) {
          float hj = td * w1r[j] + b1r[j];
          hj = hj > 0.0f ? hj : 0.0f;
          z += hj * w2r[j];
        }
        s *= 1.0f / (1.0f + __expf(-z));   // * sigmoid(mlp(|dt|))
        sv[c4] = s;
        rmax = fmaxf(rmax, s);
      }
#pragma unroll
      for (int m = 16; m >= 1; m >>= 1)
        rmax = fmaxf(rmax, __shfl_xor(rmax, m, 32));
      const float mnew  = fmaxf(mrow[rr], rmax);
      const float alpha = __expf(mrow[rr] - mnew);
      mrow[rr] = mnew;
      float psum = 0.0f;
#pragma unroll
      for (int c4 = 0; c4 < 4; ++c4) {
        const float p = __expf(sv[c4] - mnew);
        psum += p;
        sP[r * KB + c4 * 32 + lane] = (_Float16)p;
      }
#pragma unroll
      for (int m = 16; m >= 1; m >>= 1)
        psum += __shfl_xor(psum, m, 32);
      lrow[rr] = lrow[rr] * alpha + psum;
      if (lane == 0) sAlpha[r] = alpha;
    }
    // Barrier B: sP / sAlpha visible before rescale + PV.
    __syncthreads();

    // ---- 3) rescale accumulators by alpha ----
#pragma unroll
    for (int v = 0; v < 8; ++v) {
      const float a = sAlpha[v + 8 * sel];
      acc[0][v] *= a; acc[1][v] *= a; acc[2][v] *= a; acc[3][v] *= a;
    }

    // ---- 4) acc += P @ V_slice (A from LDS, B from transposed K) ----
    // All 4 B-fragments materialized before the WMMA burst per k-step.
#pragma unroll
    for (int kk = 0; kk < 4; ++kk) {           // 4 k-steps of 32 keys
      const v16h aP = ldfrag(sP + hrow * KB + kk * 32 + sel * 8);
      v16h bv[4];
#pragma unroll
      for (int t = 0; t < 4; ++t) {
        bv[t] = ldfrag(kT + (size_t)(dbase + t * 16 + hrow) * NH + kb +
                       kk * 32 + sel * 8);
      }
#pragma unroll
      for (int t = 0; t < 4; ++t) {
        acc[t] = __builtin_amdgcn_wmma_f32_16x16x32_f16(false, aP, false,
                                                        bv[t], (short)0,
                                                        acc[t], false, false);
      }
    }
  }

  // ---- epilogue: out = acc / l ----
  if (lane == 0) { sL[2 * wave] = lrow[0]; sL[2 * wave + 1] = lrow[1]; }
  __syncthreads();
#pragma unroll
  for (int v = 0; v < 8; ++v) {
    const int r = v + 8 * sel;
    const float inv = 1.0f / sL[r];
#pragma unroll
    for (int t = 0; t < 4; ++t) {
      out[(size_t)(qbase + r) * DE + dbase + t * 16 + hrow] = acc[t][v] * inv;
    }
  }
}

// ---------------------------------------------------------------------------
extern "C" void kernel_launch(void* const* d_in, const int* in_sizes, int n_in,
                              void* d_out, int out_size, void* d_ws, size_t ws_size,
                              hipStream_t stream) {
  const float* Q  = (const float*)d_in[0];  // query_code_embeds [N,D]
  const float* K  = (const float*)d_in[1];  // hist_embeddings  [N,D]
  const float* t  = (const float*)d_in[2];  // hist_times       [N]
  const float* w1 = (const float*)d_in[3];  // [1,H]
  const float* b1 = (const float*)d_in[4];  // [H]
  const float* w2 = (const float*)d_in[5];  // [H,1]
  const float* b2 = (const float*)d_in[6];  // [1]
  float* out = (float*)d_out;

  // workspace layout: q_h | k_h | kT_h   (each N*D f16 = 4 MB)
  _Float16* qh = (_Float16*)d_ws;
  _Float16* kh = qh + (size_t)NH * DE;
  _Float16* kT = kh + (size_t)NH * DE;

  const int total = NH * DE;
  taa_convert_kernel<<<total / 256, 256, 0, stream>>>(Q, K, qh, kh, kT);
  taa_attn_kernel<<<NH / QB, 256, 0, stream>>>(qh, kh, kT, t, w1, b1, w2, b2, out);
  (void)in_sizes; (void)n_in; (void)out_size; (void)ws_size;
}
